// GCNNClassifierV4_69389491634802
// MI455X (gfx1250) — matrix operand order repaired and checked by reading, not compiled, not run on hardware
//
#include <hip/hip_runtime.h>
#include <hip/hip_bf16.h>
#include <math.h>

// ---------------------------------------------------------------------------
// CDNA5 (gfx1250) WMMA types
// ---------------------------------------------------------------------------
typedef __attribute__((ext_vector_type(16))) __bf16 v16bf;
typedef __attribute__((ext_vector_type(8)))  __bf16 v8bf;
typedef __attribute__((ext_vector_type(8)))  float  v8f;

#define DEV static __device__ __forceinline__

DEV float gelu_f(float x) { return 0.5f * x * (1.0f + erff(x * 0.70710678118654752f)); }

DEV void atomicMaxF(float* addr, float v) {
  // float atomic max via signed/unsigned int ordering trick
  if (v >= 0.0f) atomicMax((int*)addr, __float_as_int(v));
  else           atomicMin((unsigned int*)addr, __float_as_uint(v));
}

DEV v16bf join16(v8bf lo, v8bf hi) {
  return __builtin_shufflevector(lo, hi, 0,1,2,3,4,5,6,7,8,9,10,11,12,13,14,15);
}

// ---------------------------------------------------------------------------
// WMMA GEMM: C[M,N] = A[M,K](bf16,row-major) * B (given as Bt[N,K] bf16) + bias
// One wave per 16-row M tile; wave sweeps groups of FOUR 16-wide N tiles
// (16x64 register tile, 4 f32x8 accumulators) so the A fragment is loaded
// once per K-step instead of once per N tile (4x less L2 traffic on A).
// Requires M%16==0, K%32==0, N%64==0 (true for every GEMM in this model).
// ---------------------------------------------------------------------------
#define GEMM_WAVES 4
__global__ __launch_bounds__(32 * GEMM_WAVES)
void wmma_gemm_kernel(const __bf16* __restrict__ A, const __bf16* __restrict__ Bt,
                      const float* __restrict__ bias, float* __restrict__ C,
                      int M, int K, int N) {
  const int wave  = threadIdx.x >> 5;
  const int lane  = threadIdx.x & 31;
  const int mTile = blockIdx.x * GEMM_WAVES + wave;
  if (mTile * 16 >= M) return;                 // whole wave exits together (EXEC stays full)
  const int laneLo = lane & 15;
  const int kh     = lane >> 4;                // which K-half this lane supplies
  const int row    = mTile * 16 + laneLo;
  const __bf16* aRow = A + (size_t)row * K + kh * 16;
  const int nGroups = N >> 6;                  // groups of 4 N-tiles (64 cols)
  for (int ng = 0; ng < nGroups; ++ng) {
    const int col0 = (ng << 6) + laneLo;
    const __bf16* bRow0 = Bt + (size_t)col0 * K + kh * 16;
    const __bf16* bRow1 = bRow0 + (size_t)16 * K;
    const __bf16* bRow2 = bRow0 + (size_t)32 * K;
    const __bf16* bRow3 = bRow0 + (size_t)48 * K;
    v8f acc0 = {0.f,0.f,0.f,0.f,0.f,0.f,0.f,0.f};
    v8f acc1 = acc0, acc2 = acc0, acc3 = acc0;
    for (int k0 = 0; k0 < K; k0 += 32) {
      // A lane data: K = k0+kh*16+[0..15] contiguous
      v16bf a = join16(*(const v8bf*)(aRow + k0), *(const v8bf*)(aRow + k0 + 8));
      if (k0 + 32 < K) __builtin_prefetch(aRow + k0 + 32, 0, 1);  // global_prefetch_b8
      // B lane data per tile: K = k0+kh*16+[0..15] contiguous in Wt row
      v16bf b0 = join16(*(const v8bf*)(bRow0 + k0), *(const v8bf*)(bRow0 + k0 + 8));
      v16bf b1 = join16(*(const v8bf*)(bRow1 + k0), *(const v8bf*)(bRow1 + k0 + 8));
      v16bf b2 = join16(*(const v8bf*)(bRow2 + k0), *(const v8bf*)(bRow2 + k0 + 8));
      v16bf b3 = join16(*(const v8bf*)(bRow3 + k0), *(const v8bf*)(bRow3 + k0 + 8));
      acc0 = __builtin_amdgcn_wmma_f32_16x16x32_bf16(false, a, false, b0, (short)0, acc0, false, false);
      acc1 = __builtin_amdgcn_wmma_f32_16x16x32_bf16(false, a, false, b1, (short)0, acc1, false, false);
      acc2 = __builtin_amdgcn_wmma_f32_16x16x32_bf16(false, a, false, b2, (short)0, acc2, false, false);
      acc3 = __builtin_amdgcn_wmma_f32_16x16x32_bf16(false, a, false, b3, (short)0, acc3, false, false);
    }
    // C layout: lane holds col (lane&15), rows rbase..rbase+7 in VGPRs 0..7
    const int rbase = mTile * 16 + kh * 8;
    v8f accs[4] = {acc0, acc1, acc2, acc3};
#pragma unroll
    for (int t = 0; t < 4; ++t) {
      const int col = col0 + t * 16;
      const float bv = bias ? bias[col] : 0.0f;
      float* cPtr = C + (size_t)rbase * N + col;
#pragma unroll
      for (int r = 0; r < 8; ++r) cPtr[(size_t)r * N] = accs[t][r] + bv;
    }
  }
}

// ---------------------------------------------------------------------------
// Converters / fills
// ---------------------------------------------------------------------------
__global__ void f32_to_bf16_kernel(const float* __restrict__ in, __bf16* __restrict__ out,
                                   size_t n) {
  size_t i = (size_t)blockIdx.x * blockDim.x + threadIdx.x;
  if (i < n) out[i] = (__bf16)in[i];
}

__global__ void transpose_bf16_kernel(const float* __restrict__ W, __bf16* __restrict__ Wt,
                                      int K, int N) {
  size_t i = (size_t)blockIdx.x * blockDim.x + threadIdx.x;
  if (i >= (size_t)K * N) return;
  int k = (int)(i / N), c = (int)(i % N);
  Wt[(size_t)c * K + k] = (__bf16)W[i];
}

__global__ void fill_kernel(float* __restrict__ p, float v, size_t n) {
  size_t i = (size_t)blockIdx.x * blockDim.x + threadIdx.x;
  if (i < n) p[i] = v;
}

// ---------------------------------------------------------------------------
// Embed: h = gelu(ln(x@W + b)); one 64-thread block per node (D=64, K=16)
// ---------------------------------------------------------------------------
__global__ __launch_bounds__(64)
void embed_kernel(const float* __restrict__ x, const float* __restrict__ W,
                  const float* __restrict__ b, const float* __restrict__ g,
                  const float* __restrict__ bb, float* __restrict__ h) {
  __shared__ float sred[64];
  __shared__ float xs[16];
  const int i = blockIdx.x, t = threadIdx.x;
  if (t < 16) xs[t] = x[(size_t)i * 16 + t];
  __syncthreads();
  float acc = b[t];
#pragma unroll
  for (int k = 0; k < 16; ++k) acc += xs[k] * W[k * 64 + t];
  sred[t] = acc; __syncthreads();
  for (int s = 32; s > 0; s >>= 1) { if (t < s) sred[t] += sred[t + s]; __syncthreads(); }
  float mean = sred[0] * (1.0f / 64.0f); __syncthreads();
  float d = acc - mean;
  sred[t] = d * d; __syncthreads();
  for (int s = 32; s > 0; s >>= 1) { if (t < s) sred[t] += sred[t + s]; __syncthreads(); }
  float var = sred[0] * (1.0f / 64.0f);
  float gg = d * __frsqrt_rn(var + 1e-5f) * g[t] + bb[t];
  h[(size_t)i * 64 + t] = gelu_f(gg);
}

// ---------------------------------------------------------------------------
// Attention logits: al_s/al_d [n,H] from xw [n,H*C]
// ---------------------------------------------------------------------------
__global__ void attn_logits_kernel(const float* __restrict__ xw, const float* __restrict__ a_src,
                                   const float* __restrict__ a_dst, float* __restrict__ als,
                                   float* __restrict__ ald, int n, int H, int C) {
  int idx = blockIdx.x * blockDim.x + threadIdx.x;
  if (idx >= n * H) return;
  int node = idx / H, hd = idx % H;
  const float* xp = xw + (size_t)node * H * C + hd * C;
  float s = 0.f, d = 0.f;
  for (int c = 0; c < C; ++c) { float v = xp[c]; s += v * a_src[hd * C + c]; d += v * a_dst[hd * C + c]; }
  als[idx] = s; ald[idx] = d;
}

// ---------------------------------------------------------------------------
// Edge softmax (3 passes). Edge list is [src row | dst row] + n self-loops.
// ---------------------------------------------------------------------------
DEV void edge_ids(const int* ei, int E, int e, int& s, int& d) {
  if (e < E) { s = ei[e]; d = ei[E + e]; } else { s = e - E; d = e - E; }
}

__global__ void edge_max_kernel(const int* __restrict__ ei, int E, int n, int H,
                                const float* __restrict__ als, const float* __restrict__ ald,
                                float* __restrict__ ebuf, float* __restrict__ emax) {
  int idx = blockIdx.x * blockDim.x + threadIdx.x;
  int Et = E + n;
  if (idx >= Et * H) return;
  int e = idx / H, hd = idx % H, s, d;
  edge_ids(ei, E, e, s, d);
  float x = als[s * H + hd] + ald[d * H + hd];
  x = x > 0.f ? x : 0.2f * x;                 // LeakyReLU(0.2)
  ebuf[idx] = x;
  atomicMaxF(&emax[d * H + hd], x);
}

__global__ void edge_exp_kernel(const int* __restrict__ ei, int E, int n, int H,
                                float* __restrict__ ebuf, const float* __restrict__ emax,
                                float* __restrict__ den) {
  int idx = blockIdx.x * blockDim.x + threadIdx.x;
  int Et = E + n;
  if (idx >= Et * H) return;
  int e = idx / H, hd = idx % H, s, d;
  edge_ids(ei, E, e, s, d);
  float ex = expf(ebuf[idx] - emax[d * H + hd]);
  ebuf[idx] = ex;
  atomicAdd(&den[d * H + hd], ex);
}

__global__ void edge_agg_kernel(const int* __restrict__ ei, int E, int n, int H, int C,
                                const float* __restrict__ ebuf, const float* __restrict__ den,
                                const float* __restrict__ xw, float* __restrict__ agg) {
  int idx = blockIdx.x * blockDim.x + threadIdx.x;
  int Et = E + n;
  if (idx >= Et * H) return;
  int e = idx / H, hd = idx % H, s, d;
  edge_ids(ei, E, e, s, d);
  float coef = ebuf[idx] / (den[d * H + hd] + 1e-16f);
  const float* xp = xw + (size_t)s * H * C + hd * C;
  float* op = agg + (size_t)d * H * C + hd * C;
  for (int c = 0; c < C; ++c) atomicAdd(&op[c], xp[c] * coef);
}

// ---------------------------------------------------------------------------
// GAT post: h = gelu( LN(agg + bias) + res ).  blockDim == D (128/256).
// agg may alias hout (per-element, per-row safe).
// ---------------------------------------------------------------------------
__global__ void gat_post_kernel(const float* __restrict__ agg, const float* __restrict__ bias,
                                const float* __restrict__ lng, const float* __restrict__ lnb,
                                const float* __restrict__ res, float* __restrict__ hout, int D) {
  __shared__ float sred[512];
  const int i = blockIdx.x, t = threadIdx.x;
  float v = agg[(size_t)i * D + t] + bias[t];
  sred[t] = v; __syncthreads();
  for (int s = D >> 1; s > 0; s >>= 1) { if (t < s) sred[t] += sred[t + s]; __syncthreads(); }
  float mean = sred[0] / (float)D; __syncthreads();
  float dd = v - mean;
  sred[t] = dd * dd; __syncthreads();
  for (int s = D >> 1; s > 0; s >>= 1) { if (t < s) sred[t] += sred[t + s]; __syncthreads(); }
  float var = sred[0] / (float)D;
  float g = dd * __frsqrt_rn(var + 1e-5f) * lng[t] + lnb[t];
  hout[(size_t)i * D + t] = gelu_f(g + res[(size_t)i * D + t]);
}

// LN + GELU, writes f32 and bf16 (bf16 feeds next WMMA GEMM). blockDim == D.
__global__ void ln_gelu_kernel(const float* __restrict__ in, const float* __restrict__ lng,
                               const float* __restrict__ lnb, float* __restrict__ outf,
                               __bf16* __restrict__ outb, int D) {
  __shared__ float sred[512];
  const int i = blockIdx.x, t = threadIdx.x;
  float v = in[(size_t)i * D + t];
  sred[t] = v; __syncthreads();
  for (int s = D >> 1; s > 0; s >>= 1) { if (t < s) sred[t] += sred[t + s]; __syncthreads(); }
  float mean = sred[0] / (float)D; __syncthreads();
  float dd = v - mean;
  sred[t] = dd * dd; __syncthreads();
  for (int s = D >> 1; s > 0; s >>= 1) { if (t < s) sred[t] += sred[t + s]; __syncthreads(); }
  float var = sred[0] / (float)D;
  float g = gelu_f(dd * __frsqrt_rn(var + 1e-5f) * lng[t] + lnb[t]);
  outf[(size_t)i * D + t] = g;
  outb[(size_t)i * D + t] = (__bf16)g;
}

__global__ void gelu_kernel(const float* __restrict__ in, float* __restrict__ out, size_t n) {
  size_t i = (size_t)blockIdx.x * blockDim.x + threadIdx.x;
  if (i < n) out[i] = gelu_f(in[i]);
}

// ---------------------------------------------------------------------------
// Graph pooling
// ---------------------------------------------------------------------------
__global__ void pool_kernel(const float* __restrict__ h, const int* __restrict__ batch,
                            int n, int D, float* __restrict__ s, float* __restrict__ mx,
                            float* __restrict__ cnt) {
  size_t idx = (size_t)blockIdx.x * blockDim.x + threadIdx.x;
  if (idx >= (size_t)n * D) return;
  int i = (int)(idx / D), c = (int)(idx % D);
  int g = batch[i];
  float v = h[idx];
  atomicAdd(&s[(size_t)g * D + c], v);
  atomicMaxF(&mx[(size_t)g * D + c], v);
  if (c == 0) atomicAdd(&cnt[g], 1.0f);
}

__global__ void make_gf_kernel(const float* __restrict__ s, const float* __restrict__ mx,
                               const float* __restrict__ cnt, const float* __restrict__ z,
                               float* __restrict__ cbuf, int G) {
  int idx = blockIdx.x * blockDim.x + threadIdx.x;
  if (idx >= G * 576) return;
  int g = idx / 576, c = idx % 576;
  float v;
  if (c < 128)       v = s[g * 128 + c] / fmaxf(cnt[g], 1.0f);
  else if (c < 256) { float m = mx[g * 128 + (c - 128)]; v = (m >= -3.0e38f) ? m : 0.0f; }
  else if (c < 384)  v = s[g * 128 + (c - 256)] * 0.1f;
  else               v = z[g * 192 + (c - 384)];
  cbuf[idx] = v;
}

__global__ void clf_out_kernel(const float* __restrict__ a, const float* __restrict__ W2,
                               const float* __restrict__ b2, float* __restrict__ out, int G) {
  int g = blockIdx.x * blockDim.x + threadIdx.x;
  if (g >= G) return;
  float acc = b2[0];
  for (int k = 0; k < 64; ++k) acc += a[(size_t)g * 64 + k] * W2[k];
  out[g] = acc;
}

// ---------------------------------------------------------------------------
// Host orchestration
// ---------------------------------------------------------------------------
static inline size_t cdiv(size_t a, size_t b) { return (a + b - 1) / b; }

extern "C" void kernel_launch(void* const* d_in, const int* in_sizes, int n_in,
                              void* d_out, int out_size, void* d_ws, size_t ws_size,
                              hipStream_t stream) {
  (void)n_in; (void)ws_size; (void)out_size;
  const int n = in_sizes[0] / 16;          // 50000 nodes (16*3125 -> M%16==0)
  const int E = in_sizes[1] / 2;           // 800000 edges
  const int G = in_sizes[3] / 64;          // 512 graphs
  const int H = 4;
  const int Et = E + n;

  const float* x     = (const float*)d_in[0];
  const int*   ei    = (const int*)d_in[1];
  const int*   batch = (const int*)d_in[2];
  const float* u     = (const float*)d_in[3];
  // params in setup_inputs() insertion order:
  // embed: 4..7 | gat1: 8..15 | gat2: 16..23 | gat3: 24..29 | gat4: 30..37
  // genc: 38..47 | fusion: 48..59 | clf: 60..63
  auto P = [&](int i) { return (const float*)d_in[i]; };

  // ---- workspace layout --------------------------------------------------
  size_t off = 0;
  auto carve = [&](size_t bytes) { void* p = (char*)d_ws + off; off += (bytes + 255) & ~(size_t)255; return p; };
  float*  h    = (float*)carve((size_t)n * 256 * 4);
  float*  xw   = (float*)carve((size_t)n * 256 * 4);
  float*  res  = (float*)carve((size_t)n * 256 * 4);
  __bf16* hbf  = (__bf16*)carve((size_t)n * 256 * 2);
  __bf16* wbf  = (__bf16*)carve((size_t)576 * 512 * 2);
  __bf16* abf  = (__bf16*)carve((size_t)G * 576 * 2);
  float*  als  = (float*)carve((size_t)n * H * 4);
  float*  ald  = (float*)carve((size_t)n * H * 4);
  float*  emax = (float*)carve((size_t)n * H * 4);
  float*  den  = (float*)carve((size_t)n * H * 4);
  float*  ebuf = (float*)carve((size_t)Et * H * 4);
  float*  ps   = (float*)carve((size_t)G * 128 * 4);
  float*  pmx  = (float*)carve((size_t)G * 128 * 4);
  float*  pcnt = (float*)carve((size_t)G * 4);
  float*  gA   = (float*)carve((size_t)G * 576 * 4);
  float*  gB   = (float*)carve((size_t)G * 576 * 4);

  auto fill = [&](float* p, float v, size_t cnt) {
    fill_kernel<<<dim3(cdiv(cnt, 256)), dim3(256), 0, stream>>>(p, v, cnt);
  };
  auto tobf = [&](const float* in, __bf16* out, size_t cnt) {
    f32_to_bf16_kernel<<<dim3(cdiv(cnt, 256)), dim3(256), 0, stream>>>(in, out, cnt);
  };
  auto wt = [&](const float* W, int K, int N) {
    transpose_bf16_kernel<<<dim3(cdiv((size_t)K * N, 256)), dim3(256), 0, stream>>>(W, wbf, K, N);
  };
  auto gemm = [&](const __bf16* A, const float* bias, float* C, int M, int K, int N) {
    wmma_gemm_kernel<<<dim3(cdiv((size_t)M / 16, GEMM_WAVES)), dim3(32 * GEMM_WAVES), 0, stream>>>(
        A, wbf, bias, C, M, K, N);
  };

  // ---- node embedding ----------------------------------------------------
  embed_kernel<<<dim3(n), dim3(64), 0, stream>>>(x, P(4), P(5), P(6), P(7), h);

  // ---- 4 GAT blocks ------------------------------------------------------
  struct L { int base, Fi, Fo; bool hasRes; };
  const L layers[4] = { {8, 64, 128, true}, {16, 128, 256, true},
                        {24, 256, 256, false}, {30, 256, 128, true} };
  const int eGrid = (int)cdiv((size_t)Et * H, 256);
  for (int li = 0; li < 4; ++li) {
    const L& L_ = layers[li];
    const int Fi = L_.Fi, Fo = L_.Fo, C = Fo / H, b = L_.base;
    tobf(h, hbf, (size_t)n * Fi);
    wt(P(b + 0), Fi, Fo);
    gemm(hbf, nullptr, xw, n, Fi, Fo);                     // xw = h @ W
    if (L_.hasRes) { wt(P(b + 6), Fi, Fo); gemm(hbf, P(b + 7), res, n, Fi, Fo); }
    attn_logits_kernel<<<dim3(cdiv((size_t)n * H, 256)), dim3(256), 0, stream>>>(
        xw, P(b + 1), P(b + 2), als, ald, n, H, C);
    float* agg = L_.hasRes ? h : res;                      // alias agg with free buffer
    const float* resSrc = L_.hasRes ? res : h;
    fill(emax, -INFINITY, (size_t)n * H);
    fill(den, 0.0f, (size_t)n * H);
    fill(agg, 0.0f, (size_t)n * Fo);
    edge_max_kernel<<<dim3(eGrid), dim3(256), 0, stream>>>(ei, E, n, H, als, ald, ebuf, emax);
    edge_exp_kernel<<<dim3(eGrid), dim3(256), 0, stream>>>(ei, E, n, H, ebuf, emax, den);
    edge_agg_kernel<<<dim3(eGrid), dim3(256), 0, stream>>>(ei, E, n, H, C, ebuf, den, xw, agg);
    gat_post_kernel<<<dim3(n), dim3(Fo), 0, stream>>>(agg, P(b + 3), P(b + 4), P(b + 5),
                                                      resSrc, h, Fo);
  }

  // ---- graph pooling (h is [n,128]) --------------------------------------
  fill(ps, 0.0f, (size_t)G * 128);
  fill(pmx, -INFINITY, (size_t)G * 128);
  fill(pcnt, 0.0f, (size_t)G);
  pool_kernel<<<dim3(cdiv((size_t)n * 128, 256)), dim3(256), 0, stream>>>(h, batch, n, 128,
                                                                          ps, pmx, pcnt);

  // ---- graph encoder: u -> z [G,192] -------------------------------------
  tobf(u, abf, (size_t)G * 64);
  wt(P(38), 64, 256);  gemm(abf, P(39), gA, G, 64, 256);
  ln_gelu_kernel<<<dim3(G), dim3(256), 0, stream>>>(gA, P(40), P(41), gA, abf, 256);
  wt(P(42), 256, 256); gemm(abf, P(43), gB, G, 256, 256);
  ln_gelu_kernel<<<dim3(G), dim3(256), 0, stream>>>(gB, P(44), P(45), gB, abf, 256);
  wt(P(46), 256, 192); gemm(abf, P(47), gA, G, 256, 192);           // z in gA

  // ---- fuse: cbuf = [mean|max|s/10|z] [G,576] -----------------------------
  make_gf_kernel<<<dim3(cdiv((size_t)G * 576, 256)), dim3(256), 0, stream>>>(ps, pmx, pcnt,
                                                                             gA, gB, G);
  tobf(gB, abf, (size_t)G * 576);
  wt(P(48), 576, 512); gemm(abf, P(49), gA, G, 576, 512);
  ln_gelu_kernel<<<dim3(G), dim3(512), 0, stream>>>(gA, P(50), P(51), gA, abf, 512);
  wt(P(52), 512, 256); gemm(abf, P(53), gB, G, 512, 256);
  ln_gelu_kernel<<<dim3(G), dim3(256), 0, stream>>>(gB, P(54), P(55), gB, abf, 256);
  wt(P(56), 256, 128); gemm(abf, P(57), gA, G, 256, 128);
  ln_gelu_kernel<<<dim3(G), dim3(128), 0, stream>>>(gA, P(58), P(59), gA, abf, 128);

  // ---- classifier head ----------------------------------------------------
  wt(P(60), 128, 64);  gemm(abf, P(61), gB, G, 128, 64);
  gelu_kernel<<<dim3(cdiv((size_t)G * 64, 256)), dim3(256), 0, stream>>>(gB, gB, (size_t)G * 64);
  clf_out_kernel<<<dim3(cdiv((size_t)G, 256)), dim3(256), 0, stream>>>(gB, P(62), P(63),
                                                                       (float*)d_out, G);
}
